// RIM_37692632990225
// MI455X (gfx1250) — matrix operand order repaired and checked by reading, not compiled, not run on hardware
//
#include <hip/hip_runtime.h>
#include <hip/hip_bf16.h>

// ---- problem constants ----
#define B_SZ   512
#define IN_SZ  1024
#define H_SZ   1024
#define U_SZ   6
#define K_TOP  4
#define IKS_SZ 64
#define IVS_SZ 512
#define NCH_SZ 4
#define CQS_SZ 32
#define CVS_SZ 100
#define LN_EPS 1e-5f

typedef __attribute__((ext_vector_type(16))) __bf16 v16bf;
typedef __attribute__((ext_vector_type(8)))  float  v8f;

__device__ __forceinline__ __bf16 to_bf16(float f) { return (__bf16)f; }

// ---------------------------------------------------------------------------
// LDS-staged, register-pipelined batched GEMM:
//   C[m,n] = rowscale[m]*sum_k A[m,k]*B[k,n] + addend[m,n] + bias1[n] + bias2[n]
// Block tile: 128(M) x (NTILES*16)(N), K-step 32, 8 waves (wave32).
// Pipeline per K-step: store prefetched regs -> LDS (f32->bf16 cvt), barrier,
// issue next tile's global loads into regs (scalar-guarded), then branch-free
// fragment assembly (ds_load_b128) + WMMA, barrier. Global-load latency of
// tile k+1 overlaps the WMMAs of tile k. All K-tail guards live in staging;
// the WMMA region never touches EXEC (§7.12 restriction).
// Fragment layouts per §7.12.2:
//   A 16x32: lane m=L%16, elem e -> k = 16*(e/8) + 8*(L/16) + (e%8)
//   B 32x16: lane n=L%16, elem e -> k = 16*(e/8) + 8*(L/16) + (e%8)
//   C 16x16: lane n=L%16, vgpr r -> m = r + 8*(L/16)
// ---------------------------------------------------------------------------
template <bool BT, int NTILES> // BT=false: B is KxN row-major; true: NxK
__global__ __launch_bounds__(256)
void gemm_wmma_bf16(const float* __restrict__ A, int lda, long long a_bs,
                    const float* __restrict__ Bm, int ldb, long long b_bs,
                    float* __restrict__ C, int ldc, long long c_bs,
                    int Kd, int n_off,
                    const float* __restrict__ bias1, long long bias_bs,
                    const float* __restrict__ bias2,
                    const float* __restrict__ rowscale, int rs_mstride,
                    long long rs_bs,
                    const float* __restrict__ addend, int ld_add,
                    long long add_bs) {
  constexpr int BM = 128, BK = 32, BN = NTILES * 16;
  constexpr int LDW = BK + 8; // LDS row stride (bf16 elems), padded
  __shared__ __bf16 As[BM][LDW];
  __shared__ __bf16 Bs[BN][LDW]; // n-major (transposed) for contiguous k reads

  const int batch = blockIdx.z;
  A += (long long)batch * a_bs;
  Bm += (long long)batch * b_bs;
  C += (long long)batch * c_bs;
  if (bias1)    bias1    += (long long)batch * bias_bs;
  if (bias2)    bias2    += (long long)batch * bias_bs;
  if (rowscale) rowscale += (long long)batch * rs_bs;
  if (addend)   addend   += (long long)batch * add_bs;

  const int tid = threadIdx.x;
  const int wave = tid >> 5; // wave32
  const int lane = tid & 31;
  const int m_blk = (int)blockIdx.y * BM;
  const int nb = n_off + (int)blockIdx.x * BN;

  // staging maps (fixed per thread)
  const int arow = tid >> 1;          // 0..127
  const int ahalf = (tid & 1) << 4;   // 0 or 16
  constexpr int NQ4 = BT ? (BN * 8) : (BK * (BN / 4)); // total float4 in B tile
  constexpr int BITER = (NQ4 + 255) / 256;

  // compute-phase fragment coordinates (fixed per lane)
  const int frow = wave * 16 + (lane & 15); // A row in LDS
  const int khalf = ((lane >> 4) << 3);     // 0 or 8
  const int kb = ((lane >> 4) << 4);        // 0 or 16 (unused)

  float4 aregs[4];
  float4 breg[BITER];

  auto loadA = [&](int k0) {
    const float* aptr = A + (long long)(m_blk + arow) * lda + k0 + ahalf;
    if (k0 + ahalf + 16 <= Kd) {
#pragma unroll
      for (int j = 0; j < 4; ++j) aregs[j] = *(const float4*)(aptr + (j << 2));
    } else {
#pragma unroll
      for (int j = 0; j < 4; ++j) {
        const int kk = k0 + ahalf + (j << 2);
        float4 f;
        f.x = (kk + 0 < Kd) ? aptr[(j << 2) + 0] : 0.0f;
        f.y = (kk + 1 < Kd) ? aptr[(j << 2) + 1] : 0.0f;
        f.z = (kk + 2 < Kd) ? aptr[(j << 2) + 2] : 0.0f;
        f.w = (kk + 3 < Kd) ? aptr[(j << 2) + 3] : 0.0f;
        aregs[j] = f;
      }
    }
  };
  auto storeA = [&]() {
#pragma unroll
    for (int j = 0; j < 4; ++j) {
      As[arow][ahalf + (j << 2) + 0] = to_bf16(aregs[j].x);
      As[arow][ahalf + (j << 2) + 1] = to_bf16(aregs[j].y);
      As[arow][ahalf + (j << 2) + 2] = to_bf16(aregs[j].z);
      As[arow][ahalf + (j << 2) + 3] = to_bf16(aregs[j].w);
    }
  };
  auto loadB = [&](int k0) {
#pragma unroll
    for (int it = 0; it < BITER; ++it) {
      const int i = tid + it * 256;
      if ((NQ4 % 256 != 0) && i >= NQ4) break;
      if (!BT) { // B[k][n]: float4 along n, transpose into LDS
        const int k = i / (BN / 4);
        const int nq = (i % (BN / 4)) << 2;
        if (k0 + k < Kd) {
          breg[it] = *(const float4*)(Bm + (long long)(k0 + k) * ldb + nb + nq);
        } else {
          breg[it].x = breg[it].y = breg[it].z = breg[it].w = 0.0f;
        }
      } else { // B[n][k]: float4 along k
        const int n = i >> 3;
        const int kq = (i & 7) << 2;
        const float* bp = Bm + (long long)(nb + n) * ldb + k0 + kq;
        if (k0 + kq + 4 <= Kd) {
          breg[it] = *(const float4*)bp;
        } else {
          float4 f;
          f.x = (k0 + kq + 0 < Kd) ? bp[0] : 0.0f;
          f.y = (k0 + kq + 1 < Kd) ? bp[1] : 0.0f;
          f.z = (k0 + kq + 2 < Kd) ? bp[2] : 0.0f;
          f.w = (k0 + kq + 3 < Kd) ? bp[3] : 0.0f;
          breg[it] = f;
        }
      }
    }
  };
  auto storeB = [&]() {
#pragma unroll
    for (int it = 0; it < BITER; ++it) {
      const int i = tid + it * 256;
      if ((NQ4 % 256 != 0) && i >= NQ4) break;
      if (!BT) {
        const int k = i / (BN / 4);
        const int nq = (i % (BN / 4)) << 2;
        Bs[nq + 0][k] = to_bf16(breg[it].x);
        Bs[nq + 1][k] = to_bf16(breg[it].y);
        Bs[nq + 2][k] = to_bf16(breg[it].z);
        Bs[nq + 3][k] = to_bf16(breg[it].w);
      } else {
        const int n = i >> 3;
        const int kq = (i & 7) << 2;
        Bs[n][kq + 0] = to_bf16(breg[it].x);
        Bs[n][kq + 1] = to_bf16(breg[it].y);
        Bs[n][kq + 2] = to_bf16(breg[it].z);
        Bs[n][kq + 3] = to_bf16(breg[it].w);
      }
    }
  };

  v8f acc[NTILES] = {};

  loadA(0);
  loadB(0);
  for (int k0 = 0; k0 < Kd; k0 += 32) {
    storeA();
    storeB();
    __syncthreads();

    // prefetch next K-tile into registers (scalar-uniform guard); its load
    // latency overlaps the WMMAs below.
    if (k0 + 32 < Kd) {
      loadA(k0 + 32);
      loadB(k0 + 32);
    }

    // ---- branch-free fragment assembly + WMMA ----
    v16bf a;
#pragma unroll
    for (int e = 0; e < 8; ++e) a[e] = As[frow][khalf + e];
#pragma unroll
    for (int e = 0; e < 8; ++e) a[8 + e] = As[frow][16 + khalf + e];

#pragma unroll
    for (int t = 0; t < NTILES; ++t) {
      const int nrel = t * 16 + (lane & 15);
      v16bf b;
#pragma unroll
      for (int e = 0; e < 16; ++e) b[e] = Bs[nrel][(e < 8) ? (khalf + e) : (16 + khalf + (e - 8))];
      acc[t] = __builtin_amdgcn_wmma_f32_16x16x32_bf16(
          false, a, false, b, (short)0, acc[t], false, false);
    }
    __syncthreads();
  }

  // ---- epilogue ----
  const int r_base = ((lane >> 4) << 3);
  const int m_wave = m_blk + wave * 16;
#pragma unroll
  for (int t = 0; t < NTILES; ++t) {
    const int n = nb + t * 16 + (lane & 15);
    float badd = 0.0f;
    if (bias1) badd += bias1[n];
    if (bias2) badd += bias2[n];
#pragma unroll
    for (int r = 0; r < 8; ++r) {
      const int m = m_wave + r_base + r;
      float v = acc[t][r];
      if (rowscale) v *= rowscale[(long long)m * rs_mstride];
      if (addend)   v += addend[(long long)m * ld_add + n];
      C[(long long)m * ldc + n] = v + badd;
    }
  }
}

// ---------------------------------------------------------------------------
// scores0[b,u] = dot(q_l[b,u,:], k0[b,:]) / 8 ; top-4-of-6 stable mask;
// alpha = sigmoid(score0) * mask   (softmax over {s0, 0} -> sigmoid)
// ---------------------------------------------------------------------------
__global__ __launch_bounds__(192)
void scores_topk_kernel(const float* __restrict__ q_l,  // (B,U,64)
                        const float* __restrict__ k0,   // (B,64)
                        float* __restrict__ alpha,      // (B,U)
                        float* __restrict__ maskout) {  // (B,U)
  __shared__ float s[U_SZ];
  const int b = blockIdx.x;
  const int wave = threadIdx.x >> 5; // == u
  const int lane = threadIdx.x & 31;
  const float* q = q_l + ((long long)b * U_SZ + wave) * IKS_SZ;
  const float* kk = k0 + (long long)b * IKS_SZ;
  float p = q[lane] * kk[lane] + q[lane + 32] * kk[lane + 32];
#pragma unroll
  for (int off = 16; off > 0; off >>= 1) p += __shfl_xor(p, off, 32);
  if (lane == 0) s[wave] = p * 0.125f; // 1/sqrt(64)
  __syncthreads();
  if (threadIdx.x < U_SZ) {
    const int u = threadIdx.x;
    const float su = s[u];
    int cnt = 0;
    for (int j = 0; j < U_SZ; ++j)
      if (s[j] > su || (s[j] == su && j < u)) ++cnt;
    const float m = (cnt < K_TOP) ? 1.0f : 0.0f;
    maskout[(long long)b * U_SZ + u] = m;
    const float p0 = 1.0f / (1.0f + __expf(-su));
    alpha[(long long)b * U_SZ + u] = p0 * m;
  }
}

// ---------------------------------------------------------------------------
// LSTM elementwise: gates (B,U,4H) -> c_new, h_bg
// ---------------------------------------------------------------------------
__global__ __launch_bounds__(256)
void lstm_kernel(const float* __restrict__ gates, const float* __restrict__ cs,
                 float* __restrict__ c_new, float* __restrict__ h_bg) {
  const long long idx = (long long)blockIdx.x * blockDim.x + threadIdx.x;
  const long long total = (long long)B_SZ * U_SZ * H_SZ;
  if (idx >= total) return;
  const long long bu = idx / H_SZ;
  const int h = (int)(idx % H_SZ);
  const float* g = gates + bu * (4 * H_SZ);
  const float gi = g[h];
  const float gf = g[H_SZ + h];
  const float gg = g[2 * H_SZ + h];
  const float go = g[3 * H_SZ + h];
  const float si = 1.0f / (1.0f + __expf(-gi));
  const float sf = 1.0f / (1.0f + __expf(-gf));
  const float so = 1.0f / (1.0f + __expf(-go));
  const float c = sf * cs[idx] + si * tanhf(gg);
  c_new[idx] = c;
  h_bg[idx] = so * tanhf(c);
}

// ---------------------------------------------------------------------------
// Communication attention: 4 heads over U=6 units, LDS-resident per batch row.
// ---------------------------------------------------------------------------
__global__ __launch_bounds__(256)
void comm_attn_kernel(const float* __restrict__ qc,    // (B,U,128)
                      const float* __restrict__ kc,    // (B,U,128)
                      const float* __restrict__ vc,    // (B,U,400)
                      const float* __restrict__ maskv, // (B,U)
                      float* __restrict__ ctx) {       // (B,U,400)
  __shared__ float q_s[U_SZ * 128];
  __shared__ float k_s[U_SZ * 128];
  __shared__ float v_s[U_SZ * 400];
  __shared__ float p_s[NCH_SZ * U_SZ * U_SZ];
  const int b = blockIdx.x;
  const int t = threadIdx.x;
  for (int i = t; i < U_SZ * 128; i += 256) {
    q_s[i] = qc[(long long)b * U_SZ * 128 + i];
    k_s[i] = kc[(long long)b * U_SZ * 128 + i];
  }
  for (int i = t; i < U_SZ * 400; i += 256)
    v_s[i] = vc[(long long)b * U_SZ * 400 + i];
  __syncthreads();
  if (t < NCH_SZ * U_SZ * U_SZ) { // 144 score entries
    const int n = t / (U_SZ * U_SZ);
    const int u = (t / U_SZ) % U_SZ;
    const int v = t % U_SZ;
    float s = 0.0f;
#pragma unroll
    for (int j = 0; j < CQS_SZ; ++j)
      s += q_s[u * 128 + n * CQS_SZ + j] * k_s[v * 128 + n * CQS_SZ + j];
    p_s[t] = s * 0.17677669529663687f; // 1/sqrt(32)
  }
  __syncthreads();
  if (t < NCH_SZ * U_SZ) { // softmax per (n,u), then query-side mask
    const int n = t / U_SZ, u = t % U_SZ;
    float* row = &p_s[(n * U_SZ + u) * U_SZ];
    float mx = row[0];
    for (int v = 1; v < U_SZ; ++v) mx = fmaxf(mx, row[v]);
    float sum = 0.0f;
    for (int v = 0; v < U_SZ; ++v) { row[v] = __expf(row[v] - mx); sum += row[v]; }
    const float inv = maskv[(long long)b * U_SZ + u] / sum;
    for (int v = 0; v < U_SZ; ++v) row[v] *= inv;
  }
  __syncthreads();
  for (int i = t; i < U_SZ * 400; i += 256) {
    const int u = i / 400, r = i % 400;
    const int n = r / CVS_SZ;
    float s = 0.0f;
#pragma unroll
    for (int v = 0; v < U_SZ; ++v)
      s += p_s[(n * U_SZ + u) * U_SZ + v] * v_s[v * 400 + r];
    ctx[(long long)b * U_SZ * 400 + i] = s;
  }
}

// ---------------------------------------------------------------------------
// LayerNorm over H + masked blend; writes hs_new then cs_new into d_out.
// ---------------------------------------------------------------------------
__global__ __launch_bounds__(256)
void ln_blend_kernel(const float* __restrict__ ctx2,  // (B,U,H)
                     const float* __restrict__ hs, const float* __restrict__ cs,
                     const float* __restrict__ c_new,
                     const float* __restrict__ maskv,
                     const float* __restrict__ ln_g,
                     const float* __restrict__ ln_b,
                     float* __restrict__ out) {
  __shared__ float red0[8];
  __shared__ float red1[8];
  const long long bu = blockIdx.x;
  const float* row = ctx2 + bu * H_SZ;
  const int t = threadIdx.x;
  float sum = 0.0f, sumsq = 0.0f;
  for (int h = t; h < H_SZ; h += 256) {
    const float v = row[h];
    sum += v;
    sumsq += v * v;
  }
#pragma unroll
  for (int off = 16; off > 0; off >>= 1) {
    sum += __shfl_xor(sum, off, 32);
    sumsq += __shfl_xor(sumsq, off, 32);
  }
  const int wave = t >> 5, lane = t & 31;
  if (lane == 0) { red0[wave] = sum; red1[wave] = sumsq; }
  __syncthreads();
  if (t == 0) {
    float a = 0.0f, q = 0.0f;
    for (int w = 0; w < 8; ++w) { a += red0[w]; q += red1[w]; }
    red0[0] = a; red1[0] = q;
  }
  __syncthreads();
  const float mu = red0[0] * (1.0f / H_SZ);
  const float var = red1[0] * (1.0f / H_SZ) - mu * mu;
  const float rstd = rsqrtf(var + LN_EPS);
  const float m = maskv[bu];
  const long long off2 = (long long)B_SZ * U_SZ * H_SZ;
  for (int h = t; h < H_SZ; h += 256) {
    const float hl = (row[h] - mu) * rstd * ln_g[h] + ln_b[h];
    out[bu * H_SZ + h] = m * hl + (1.0f - m) * hs[bu * H_SZ + h];
    out[off2 + bu * H_SZ + h] =
        m * c_new[bu * H_SZ + h] + (1.0f - m) * cs[bu * H_SZ + h];
  }
}

// ---------------------------------------------------------------------------
// host side
// ---------------------------------------------------------------------------
static inline void launch_gemm(bool btrans, int ntiles, dim3 grid,
    hipStream_t s,
    const float* A, int lda, long long abs_,
    const float* Bm, int ldb, long long bbs,
    float* C, int ldc, long long cbs, int K, int n_off,
    const float* bias1 = nullptr, long long biasbs = 0,
    const float* bias2 = nullptr,
    const float* rowscale = nullptr, int rsm = 0, long long rsbs = 0,
    const float* addend = nullptr, int ldadd = 0, long long addbs = 0) {
  if (btrans)
    gemm_wmma_bf16<true, 4><<<grid, 256, 0, s>>>(A, lda, abs_, Bm, ldb, bbs,
        C, ldc, cbs, K, n_off, bias1, biasbs, bias2, rowscale, rsm, rsbs,
        addend, ldadd, addbs);
  else if (ntiles == 4)
    gemm_wmma_bf16<false, 4><<<grid, 256, 0, s>>>(A, lda, abs_, Bm, ldb, bbs,
        C, ldc, cbs, K, n_off, bias1, biasbs, bias2, rowscale, rsm, rsbs,
        addend, ldadd, addbs);
  else
    gemm_wmma_bf16<false, 1><<<grid, 256, 0, s>>>(A, lda, abs_, Bm, ldb, bbs,
        C, ldc, cbs, K, n_off, bias1, biasbs, bias2, rowscale, rsm, rsbs,
        addend, ldadd, addbs);
}

extern "C" void kernel_launch(void* const* d_in, const int* in_sizes, int n_in,
                              void* d_out, int out_size, void* d_ws,
                              size_t ws_size, hipStream_t stream) {
  const float* x     = (const float*)d_in[0];
  const float* hs    = (const float*)d_in[1];
  const float* cs    = (const float*)d_in[2];
  const float* W_key = (const float*)d_in[3];
  const float* b_key = (const float*)d_in[4];
  const float* W_val = (const float*)d_in[5];
  const float* b_val = (const float*)d_in[6];
  const float* Wq_in = (const float*)d_in[7];
  const float* Wq_c  = (const float*)d_in[8];
  const float* Wk_c  = (const float*)d_in[9];
  const float* Wv_c  = (const float*)d_in[10];
  const float* W_co  = (const float*)d_in[11];
  const float* W_ih  = (const float*)d_in[12];
  const float* W_hh  = (const float*)d_in[13];
  const float* b_ih  = (const float*)d_in[14];
  const float* b_hh  = (const float*)d_in[15];
  const float* ln_g  = (const float*)d_in[16];
  const float* ln_b  = (const float*)d_in[17];
  float* out = (float*)d_out;
  float* ws = (float*)d_ws;

  // workspace carve (floats)
  float* k0_ws    = ws;                                      // B*64
  float* v0_ws    = k0_ws + (long long)B_SZ * IKS_SZ;        // B*512
  float* ql_ws    = v0_ws + (long long)B_SZ * IVS_SZ;        // B*U*64
  float* alpha_ws = ql_ws + (long long)B_SZ * U_SZ * IKS_SZ; // B*U
  float* mask_ws  = alpha_ws + (long long)B_SZ * U_SZ;       // B*U
  float* gates_ws = mask_ws + (long long)B_SZ * U_SZ;        // B*U*4H
  float* cnew_ws  = gates_ws + (long long)B_SZ * U_SZ * 4 * H_SZ; // B*U*H
  float* hbg_ws   = cnew_ws + (long long)B_SZ * U_SZ * H_SZ;      // B*U*H
  float* qc_ws    = hbg_ws + (long long)B_SZ * U_SZ * H_SZ;       // B*U*128
  float* kc_ws    = qc_ws + (long long)B_SZ * U_SZ * 128;         // B*U*128
  float* vc_ws    = kc_ws + (long long)B_SZ * U_SZ * 128;         // B*U*400
  float* ctx_ws   = vc_ws + (long long)B_SZ * U_SZ * 400;         // B*U*400
  float* ctx2_ws  = ctx_ws + (long long)B_SZ * U_SZ * 400;        // B*U*H

  const int gy = B_SZ / 128; // 4

  // 1) k0 = x @ W_key + b_key               (512x64, K=1024)
  launch_gemm(false, 4, dim3(1, gy, 1), stream, x, IN_SZ, 0, W_key, IKS_SZ, 0,
              k0_ws, IKS_SZ, 0, IN_SZ, 0, b_key, 0);
  // 2) v0 = x @ W_val + b_val               (512x512, K=1024)
  launch_gemm(false, 4, dim3(8, gy, 1), stream, x, IN_SZ, 0, W_val, IVS_SZ, 0,
              v0_ws, IVS_SZ, 0, IN_SZ, 0, b_val, 0);
  // 3) q_l[u] = hs[:,u] @ Wq_in[u]          (512x64, K=1024, batch=U)
  launch_gemm(false, 4, dim3(1, gy, U_SZ), stream, hs, U_SZ * H_SZ, H_SZ,
              Wq_in, IKS_SZ, (long long)H_SZ * IKS_SZ, ql_ws, U_SZ * IKS_SZ,
              IKS_SZ, IN_SZ, 0);
  // 4) scores / top-k mask / alpha
  scores_topk_kernel<<<B_SZ, 192, 0, stream>>>(ql_ws, k0_ws, alpha_ws, mask_ws);
  // 5) gates = alpha .* (v0 @ W_ih[u]^T)    (512x4096, K=512)
  launch_gemm(true, 4, dim3(64, gy, U_SZ), stream, v0_ws, IVS_SZ, 0, W_ih,
              IVS_SZ, (long long)4 * H_SZ * IVS_SZ, gates_ws, U_SZ * 4 * H_SZ,
              4 * H_SZ, IVS_SZ, 0, nullptr, 0, nullptr, alpha_ws, U_SZ, 1);
  // 6) gates += hs[:,u] @ W_hh[u]^T + b_ih + b_hh   (512x4096, K=1024)
  launch_gemm(true, 4, dim3(64, gy, U_SZ), stream, hs, U_SZ * H_SZ, H_SZ, W_hh,
              H_SZ, (long long)4 * H_SZ * H_SZ, gates_ws, U_SZ * 4 * H_SZ,
              4 * H_SZ, H_SZ, 0, b_ih, 4 * H_SZ, b_hh, nullptr, 0, 0, gates_ws,
              U_SZ * 4 * H_SZ, 4 * H_SZ);
  // 7) LSTM elementwise
  {
    const long long total = (long long)B_SZ * U_SZ * H_SZ;
    lstm_kernel<<<(unsigned)((total + 255) / 256), 256, 0, stream>>>(
        gates_ws, cs, cnew_ws, hbg_ws);
  }
  // 8) comm projections q/k/v (batch=U)
  launch_gemm(false, 4, dim3(2, gy, U_SZ), stream, hbg_ws, U_SZ * H_SZ, H_SZ,
              Wq_c, NCH_SZ * CQS_SZ, (long long)H_SZ * NCH_SZ * CQS_SZ, qc_ws,
              U_SZ * 128, 128, H_SZ, 0);
  launch_gemm(false, 4, dim3(2, gy, U_SZ), stream, hbg_ws, U_SZ * H_SZ, H_SZ,
              Wk_c, NCH_SZ * CQS_SZ, (long long)H_SZ * NCH_SZ * CQS_SZ, kc_ws,
              U_SZ * 128, 128, H_SZ, 0);
  // vc: N=400 = 6*64 + 16 -> full launch + 1-tile remainder launch
  launch_gemm(false, 4, dim3(6, gy, U_SZ), stream, hbg_ws, U_SZ * H_SZ, H_SZ,
              Wv_c, NCH_SZ * CVS_SZ, (long long)H_SZ * NCH_SZ * CVS_SZ, vc_ws,
              U_SZ * 400, 400, H_SZ, 0);
  launch_gemm(false, 1, dim3(1, gy, U_SZ), stream, hbg_ws, U_SZ * H_SZ, H_SZ,
              Wv_c, NCH_SZ * CVS_SZ, (long long)H_SZ * NCH_SZ * CVS_SZ, vc_ws,
              U_SZ * 400, 400, H_SZ, 384);
  // 9) tiny 4-head attention across units
  comm_attn_kernel<<<B_SZ, 256, 0, stream>>>(qc_ws, kc_ws, vc_ws, mask_ws,
                                             ctx_ws);
  // 10) ctx2 = ctx @ W_co[u] + h_bg        (512x1024, K=400: staged tail)
  launch_gemm(false, 4, dim3(16, gy, U_SZ), stream, ctx_ws, U_SZ * 400, 400,
              W_co, H_SZ, (long long)400 * H_SZ, ctx2_ws, U_SZ * H_SZ, H_SZ,
              400, 0, nullptr, 0, nullptr, nullptr, 0, 0, hbg_ws, U_SZ * H_SZ,
              H_SZ);
  // 11) LayerNorm + masked blend -> d_out (hs_new, cs_new)
  ln_blend_kernel<<<B_SZ * U_SZ, 256, 0, stream>>>(ctx2_ws, hs, cs, cnew_ws,
                                                   mask_ws, ln_g, ln_b, out);
}
